// ConvocationV4_2284922601700
// MI455X (gfx1250) — matrix operand order repaired and checked by reading, not compiled, not run on hardware
//
#include <hip/hip_runtime.h>
#include <hip/hip_bf16.h>
#include <math.h>

// ---------------------------------------------------------------------------
// Problem sizes (fixed by the reference)
// ---------------------------------------------------------------------------
#define BB   16
#define CC   384
#define HH   48
#define WW   48
#define HW   (HH * WW)        // 2304
#define KK   3
#define PP   (KK * KK)        // 9

// ---------------------------------------------------------------------------
// WMMA vector types (gfx1250, wave32)
// ---------------------------------------------------------------------------
typedef __attribute__((ext_vector_type(16))) __bf16 v16bf;
typedef __attribute__((ext_vector_type(8)))  float  v8f;

union BF16Frag {           // 32 bytes = one 16x32 (A) or 32x16 (B) bf16 fragment / lane
    uint4  q[2];
    v16bf  v;
};
union U16x8 {              // 16-byte staging unit for the LDS transpose
    uint4          q;
    unsigned short s[8];
};

// round-to-nearest-even f32 -> bf16 split: x ~= hi + lo (each bf16)
__device__ __forceinline__ void split_bf16(float x, unsigned short& h, unsigned short& l) {
    unsigned u  = __float_as_uint(x);
    unsigned r  = u + 0x7FFFu + ((u >> 16) & 1u);
    unsigned short hb = (unsigned short)(r >> 16);
    float hf = __uint_as_float((unsigned)hb << 16);
    float lo = x - hf;
    unsigned ul = __float_as_uint(lo);
    unsigned rl = ul + 0x7FFFu + ((ul >> 16) & 1u);
    h = hb;
    l = (unsigned short)(rl >> 16);
}

// ---------------------------------------------------------------------------
// 1) elementwise f32 -> (bf16 hi, bf16 lo)
// ---------------------------------------------------------------------------
__global__ __launch_bounds__(256) void k_split(const float* __restrict__ in,
                                               unsigned short* __restrict__ hi,
                                               unsigned short* __restrict__ lo,
                                               int n) {
    int i = blockIdx.x * 256 + threadIdx.x;
    if (i < n) {
        unsigned short h, l;
        split_bf16(in[i], h, l);
        hi[i] = h;
        lo[i] = l;
    }
}

// ---------------------------------------------------------------------------
// 2) per-(b,c) plane: 3x3 adaptive-avg-pool block means + global mean
// ---------------------------------------------------------------------------
__global__ __launch_bounds__(256) void k_pool(const float* __restrict__ x,
                                              float* __restrict__ qpool,  // [B*C, 9]
                                              float* __restrict__ emb) {  // [B*C]
    const int bc = blockIdx.x;
    const int t  = threadIdx.x;
    const float* plane = x + (size_t)bc * HW;
    __shared__ float ps[PP];
    if (t < PP) ps[t] = 0.0f;
    __syncthreads();
#pragma unroll
    for (int j = 0; j < 9; ++j) {
        int idx = t + j * 256;          // < 2304
        int h = idx / WW, w = idx - h * WW;
        int p = (h / 16) * KK + (w / 16);
        atomicAdd(&ps[p], plane[idx]);
    }
    __syncthreads();
    if (t < PP) qpool[(size_t)bc * PP + t] = ps[t] * (1.0f / 256.0f);
    if (t == 0) {
        float s = 0.0f;
#pragma unroll
        for (int p = 0; p < PP; ++p) s += ps[p];
        emb[bc] = s * (1.0f / (float)HW);
    }
}

// ---------------------------------------------------------------------------
// 3) kernel generation (tiny): one thread per (b, o)
// ---------------------------------------------------------------------------
__global__ __launch_bounds__(128) void k_genkern(const float* __restrict__ qpool, // [B*C,9]
                                                 const float* __restrict__ emb,   // [B*C]
                                                 const float* __restrict__ Wq, const float* __restrict__ bq,
                                                 const float* __restrict__ Wk, const float* __restrict__ bk,
                                                 const float* __restrict__ Wg, const float* __restrict__ bg,
                                                 float* __restrict__ kern) {      // [B*C,9]
    int idx = blockIdx.x * 128 + threadIdx.x;
    if (idx >= BB * CC) return;
    int b = idx / CC, o = idx - b * CC;

    float qk[PP];
#pragma unroll
    for (int p = 0; p < PP; ++p) qk[p] = bq[o];
    float kk = bk[o];

    const float* qpb = qpool + (size_t)b * CC * PP;
    const float* eb  = emb + (size_t)b * CC;
    for (int i = 0; i < CC; ++i) {
        float wq = Wq[(size_t)o * CC + i];
        float wk = Wk[(size_t)o * CC + i];
        kk += wk * eb[i];
        const float* qp = qpb + (size_t)i * PP;
#pragma unroll
        for (int p = 0; p < PP; ++p) qk[p] += wq * qp[p];
    }
    float g[PP], mean = 0.0f;
#pragma unroll
    for (int q = 0; q < PP; ++q) {
        float a = bg[q];
#pragma unroll
        for (int p = 0; p < PP; ++p) a += Wg[q * PP + p] * qk[p];
        g[q] = a;
        mean += a;
    }
    mean *= (1.0f / (float)PP);
    float fac = 1.0f / (1.0f + __expf(-kk));
#pragma unroll
    for (int q = 0; q < PP; ++q)
        kern[(size_t)idx * PP + q] = g[q] - fac * mean;
}

// ---------------------------------------------------------------------------
// 4) split-bf16 WMMA GEMM:  out[b][o][n] = bias[o] + sum_i A[o][i] * B[b][i][n]
//    128x128 tile / 256-thread block (8 waves, 4x2), wave = 32x64 = 2x4 frags,
//    3 WMMAs per fragment per 32-deep k-step (hi*hi + hi*lo + lo*hi).
//    Ping-pong double-buffered LDS: one barrier per k-step; next tile's global
//    loads are issued at the top of compute so the 24 WMMAs hide their latency.
// ---------------------------------------------------------------------------
#define LDA 40   // padded LDS row stride (bf16 elems): 80B rows -> conflict-free b128

__global__ __launch_bounds__(256) void k_gemm_split(
        const unsigned short* __restrict__ Ahi, const unsigned short* __restrict__ Alo,
        const unsigned short* __restrict__ Bhi, const unsigned short* __restrict__ Blo,
        const float* __restrict__ bias, float* __restrict__ out) {
    __shared__ unsigned short sAh[2][128 * LDA], sAl[2][128 * LDA];
    __shared__ unsigned short sBh[2][128 * LDA], sBl[2][128 * LDA];

    const int t     = threadIdx.x;
    const int nbase = blockIdx.x * 128;           // over 2304
    const int mbase = blockIdx.y * 128;           // over 384
    const size_t bofs = (size_t)blockIdx.z * CC * HW;

    const int lane = t & 31, wave = t >> 5;
    const int wm = wave & 3, wn = wave >> 2;      // 4x2 wave grid
    const int lrow = lane & 15, hsel = lane >> 4; // fragment lane decomposition

    // staging thread mappings (fixed per thread)
    const int arow = t >> 1, akh = (t & 1) * 16;          // A: 2 thr/row, 16 elems each
    const int bkp  = t >> 4, bnh = (t & 15) * 8;          // B: k-row pair (2k,2k+1), 8 n
    const unsigned short* aSrcH = Ahi + (size_t)(mbase + arow) * CC + akh;
    const unsigned short* aSrcL = Alo + (size_t)(mbase + arow) * CC + akh;
    const unsigned short* bSrcH = Bhi + bofs + (size_t)(2 * bkp) * HW + nbase + bnh;
    const unsigned short* bSrcL = Blo + bofs + (size_t)(2 * bkp) * HW + nbase + bnh;

    v8f acc[2][4] = {};

    auto stage = [&](int k0, int buf) {
        // ---- A tile [128 rows][32 k] (hi & lo), contiguous b128 copies ----
        {
            const uint4* sh = (const uint4*)(aSrcH + k0);
            const uint4* sl = (const uint4*)(aSrcL + k0);
            uint4* dh = (uint4*)&sAh[buf][arow * LDA + akh];
            uint4* dl = (uint4*)&sAl[buf][arow * LDA + akh];
            uint4 h0 = sh[0], h1 = sh[1], l0 = sl[0], l1 = sl[1];
            dh[0] = h0; dh[1] = h1;
            dl[0] = l0; dl[1] = l1;
        }
        // ---- B tile transposed -> LDS[n][k]; pack k-pairs into b32 stores ----
        {
            U16x8 h0, h1, l0, l1;
            const unsigned short* sh = bSrcH + (size_t)k0 * HW;
            const unsigned short* sl = bSrcL + (size_t)k0 * HW;
            h0.q = *(const uint4*)sh;
            h1.q = *(const uint4*)(sh + HW);
            l0.q = *(const uint4*)sl;
            l1.q = *(const uint4*)(sl + HW);
            __builtin_prefetch(sh + 32 * HW, 0, 0);   // gfx1250 global_prefetch
            __builtin_prefetch(sl + 32 * HW, 0, 0);
#pragma unroll
            for (int j = 0; j < 8; ++j) {
                unsigned ph = (unsigned)h0.s[j] | ((unsigned)h1.s[j] << 16);
                unsigned pl = (unsigned)l0.s[j] | ((unsigned)l1.s[j] << 16);
                *(unsigned*)&sBh[buf][(bnh + j) * LDA + 2 * bkp] = ph;
                *(unsigned*)&sBl[buf][(bnh + j) * LDA + 2 * bkp] = pl;
            }
        }
    };

    const int ksteps = CC / 32;                   // 12
    stage(0, 0);                                  // prologue

    for (int ks = 0; ks < ksteps; ++ks) {
        const int buf = ks & 1;
        __syncthreads();                          // buf stores visible; other buf free
        if (ks + 1 < ksteps) stage((ks + 1) * 32, buf ^ 1);

        // ---- fragment loads + 24 WMMAs per wave ----
        BF16Frag ah[2], al[2];
#pragma unroll
        for (int mf = 0; mf < 2; ++mf) {
            int rb = (wm * 32 + mf * 16 + lrow) * LDA + hsel * 16;  // A: 16 contiguous K per lane
            ah[mf].q[0] = *(const uint4*)&sAh[buf][rb];
            ah[mf].q[1] = *(const uint4*)&sAh[buf][rb + 8];
            al[mf].q[0] = *(const uint4*)&sAl[buf][rb];
            al[mf].q[1] = *(const uint4*)&sAl[buf][rb + 8];
        }
#pragma unroll
        for (int nf = 0; nf < 4; ++nf) {
            int cb = (wn * 64 + nf * 16 + lrow) * LDA + hsel * 16;  // B: 16 contiguous K per lane
            BF16Frag bh, bl;
            bh.q[0] = *(const uint4*)&sBh[buf][cb];  bh.q[1] = *(const uint4*)&sBh[buf][cb + 8];
            bl.q[0] = *(const uint4*)&sBl[buf][cb];  bl.q[1] = *(const uint4*)&sBl[buf][cb + 8];
#pragma unroll
            for (int mf = 0; mf < 2; ++mf) {
                acc[mf][nf] = __builtin_amdgcn_wmma_f32_16x16x32_bf16(
                    false, al[mf].v, false, bh.v, (short)0, acc[mf][nf], false, false);
                acc[mf][nf] = __builtin_amdgcn_wmma_f32_16x16x32_bf16(
                    false, ah[mf].v, false, bl.v, (short)0, acc[mf][nf], false, false);
                acc[mf][nf] = __builtin_amdgcn_wmma_f32_16x16x32_bf16(
                    false, ah[mf].v, false, bh.v, (short)0, acc[mf][nf], false, false);
            }
        }
    }

    // ---- epilogue: C/D layout -> lane holds col lrow, rows r(+8 if hsel) ----
#pragma unroll
    for (int mf = 0; mf < 2; ++mf)
#pragma unroll
        for (int nf = 0; nf < 4; ++nf) {
            int n = nbase + wn * 64 + nf * 16 + lrow;
#pragma unroll
            for (int r = 0; r < 8; ++r) {
                int o = mbase + wm * 32 + mf * 16 + hsel * 8 + r;
                out[bofs + (size_t)o * HW + n] = acc[mf][nf][r] + bias[o];
            }
        }
}

// ---------------------------------------------------------------------------
// 5) dynamic depthwise 3x3 conv (per-(b,c) kernel), 'same' zero pad,
//    output emitted directly as bf16 hi/lo for the final WMMA GEMM.
// ---------------------------------------------------------------------------
__global__ __launch_bounds__(256) void k_depthwise(const float* __restrict__ value,
                                                   const float* __restrict__ kern, // [B*C,9]
                                                   unsigned short* __restrict__ dwhi,
                                                   unsigned short* __restrict__ dwlo) {
    const int bc = blockIdx.x;
    const int t  = threadIdx.x;
    const float* plane = value + (size_t)bc * HW;
    __shared__ float tile[50 * 50];
    for (int i = t; i < 50 * 50; i += 256) tile[i] = 0.0f;
    __syncthreads();
#pragma unroll
    for (int j = 0; j < 9; ++j) {
        int idx = t + j * 256;
        int h = idx / WW, w = idx - h * WW;
        tile[(h + 1) * 50 + (w + 1)] = plane[idx];
    }
    float kw[PP];
#pragma unroll
    for (int p = 0; p < PP; ++p) kw[p] = kern[(size_t)bc * PP + p];
    __syncthreads();
#pragma unroll
    for (int j = 0; j < 9; ++j) {
        int idx = t + j * 256;
        int h = idx / WW, w = idx - h * WW;
        float acc = 0.0f;
#pragma unroll
        for (int di = 0; di < KK; ++di)
#pragma unroll
            for (int dj = 0; dj < KK; ++dj)
                acc += tile[(h + di) * 50 + (w + dj)] * kw[di * KK + dj];
        unsigned short hbits, lbits;
        split_bf16(acc, hbits, lbits);
        dwhi[(size_t)bc * HW + idx] = hbits;
        dwlo[(size_t)bc * HW + idx] = lbits;
    }
}

// ---------------------------------------------------------------------------
// launcher
// ---------------------------------------------------------------------------
extern "C" void kernel_launch(void* const* d_in, const int* in_sizes, int n_in,
                              void* d_out, int out_size, void* d_ws, size_t ws_size,
                              hipStream_t stream) {
    (void)in_sizes; (void)n_in; (void)out_size; (void)ws_size;
    const float* x  = (const float*)d_in[0];
    const float* Wq = (const float*)d_in[1];
    const float* bq = (const float*)d_in[2];
    const float* Wk = (const float*)d_in[3];
    const float* bk = (const float*)d_in[4];
    const float* Wv = (const float*)d_in[5];
    const float* bv = (const float*)d_in[6];
    const float* Wg = (const float*)d_in[7];
    const float* bg = (const float*)d_in[8];
    const float* Wp = (const float*)d_in[9];
    const float* bp = (const float*)d_in[10];
    float* out = (float*)d_out;

    // workspace carve-up (all region sizes are multiples of 256B); ~171.6 MB total
    const size_t nX  = (size_t)BB * CC * HW;     // 14,155,776
    const size_t nW  = (size_t)CC * CC;          // 147,456
    unsigned char* p = (unsigned char*)d_ws;
    unsigned short* xhi  = (unsigned short*)p; p += nX * 2;
    unsigned short* xlo  = (unsigned short*)p; p += nX * 2;
    float*          val  = (float*)p;          p += nX * 4;
    unsigned short* dwhi = (unsigned short*)p; p += nX * 2;
    unsigned short* dwlo = (unsigned short*)p; p += nX * 2;
    unsigned short* wvhi = (unsigned short*)p; p += nW * 2;
    unsigned short* wvlo = (unsigned short*)p; p += nW * 2;
    unsigned short* wphi = (unsigned short*)p; p += nW * 2;
    unsigned short* wplo = (unsigned short*)p; p += nW * 2;
    float*          qpl  = (float*)p;          p += (size_t)BB * CC * PP * 4;
    float*          emb  = (float*)p;          p += (size_t)BB * CC * 4;
    float*          kern = (float*)p;          // + B*C*9*4

    // 1) split x and the two big weight matrices into bf16 hi/lo
    k_split<<<(int)((nX + 255) / 256), 256, 0, stream>>>(x,  xhi, xlo, (int)nX);
    k_split<<<(int)((nW + 255) / 256), 256, 0, stream>>>(Wv, wvhi, wvlo, (int)nW);
    k_split<<<(int)((nW + 255) / 256), 256, 0, stream>>>(Wp, wphi, wplo, (int)nW);

    // 2) pooled statistics (block means + global mean)
    k_pool<<<BB * CC, 256, 0, stream>>>(x, qpl, emb);

    // 3) per-(b,c) dynamic 3x3 kernels
    k_genkern<<<(BB * CC + 127) / 128, 128, 0, stream>>>(qpl, emb, Wq, bq, Wk, bk, Wg, bg, kern);

    // 4) value = Wv @ x + bv      (WMMA, split-bf16)
    dim3 gg(HW / 128, CC / 128, BB);   // (18, 3, 16)
    k_gemm_split<<<gg, 256, 0, stream>>>(wvhi, wvlo, xhi, xlo, bv, val);

    // 5) dynamic depthwise conv -> bf16 hi/lo
    k_depthwise<<<BB * CC, 256, 0, stream>>>(val, kern, dwhi, dwlo);

    // 6) out = Wp @ dw + bp       (WMMA, split-bf16)
    k_gemm_split<<<gg, 256, 0, stream>>>(wphi, wplo, dwhi, dwlo, bp, out);
}
